// Attention_55894704390617
// MI455X (gfx1250) — matrix operand order repaired and checked by reading, not compiled, not run on hardware
//
#include <hip/hip_runtime.h>
#include <math.h>

#define DIMD   2048
#define SEQ    2048
#define BATCH  2
#define NHEADS 32
#define DHEAD  64
#define ROWS   (BATCH * SEQ)   // 4096

typedef _Float16 f16;
typedef __attribute__((ext_vector_type(8)))  _Float16 v8h;
typedef __attribute__((ext_vector_type(16))) _Float16 v16h;
typedef __attribute__((ext_vector_type(8)))  float    v8f;

static __device__ __forceinline__ v16h cat16(v8h lo, v8h hi) {
    v16h r;
#pragma unroll
    for (int i = 0; i < 8; ++i) { r[i] = lo[i]; r[8 + i] = hi[i]; }
    return r;
}

static __device__ __forceinline__ v8f wmma16(v16h a, v16h b, v8f c) {
    // D = A(16x32 f16) * B(32x16 f16) + C(16x16 f32)
    return __builtin_amdgcn_wmma_f32_16x16x32_f16(
        false, a, false, b, (short)0, c, false, false);
}

// ---------------------------------------------------------------------------
// 1) LayerNorm over last dim of x [ROWS, DIMD] -> xn (f16)
// ---------------------------------------------------------------------------
__global__ __launch_bounds__(256) void ln_x_kernel(const float* __restrict__ x,
                                                   const float* __restrict__ w,
                                                   f16* __restrict__ xn) {
    __shared__ float s_sum[256];
    __shared__ float s_sq[256];
    const int row = blockIdx.x;
    const int tid = threadIdx.x;
    const size_t base = (size_t)row * DIMD;

    float s = 0.f, s2 = 0.f;
#pragma unroll
    for (int i = 0; i < DIMD / 256; ++i) {
        float v = x[base + tid + i * 256];
        s += v; s2 += v * v;
    }
    s_sum[tid] = s; s_sq[tid] = s2;
    __syncthreads();
    for (int off = 128; off > 0; off >>= 1) {
        if (tid < off) { s_sum[tid] += s_sum[tid + off]; s_sq[tid] += s_sq[tid + off]; }
        __syncthreads();
    }
    const float mean = s_sum[0] * (1.0f / DIMD);
    const float var  = s_sq[0] * (1.0f / DIMD) - mean * mean;
    const float rstd = rsqrtf(var + 1e-5f);
#pragma unroll
    for (int i = 0; i < DIMD / 256; ++i) {
        int c = tid + i * 256;
        xn[base + c] = (f16)((x[base + c] - mean) * rstd * w[c]);
    }
}

// ---------------------------------------------------------------------------
// 2) Weight convert + transpose, LDS-tiled (coalesced both sides):
//    W f32 [K=DIMD, N=DIMD] -> WT f16 [N, K]
//    grid = (DIMD/32, DIMD/32), block = 256 (32x8)
// ---------------------------------------------------------------------------
__global__ __launch_bounds__(256) void wconv_kernel(const float* __restrict__ W,
                                                    f16* __restrict__ WT) {
    __shared__ f16 tile[32][33];            // +1 pad: no bank conflicts
    const int tx = threadIdx.x & 31;
    const int ty = threadIdx.x >> 5;        // 0..7
    const int k0 = blockIdx.x << 5;         // K-tile base (rows of W)
    const int n0 = blockIdx.y << 5;         // N-tile base (cols of W)
#pragma unroll
    for (int i = 0; i < 4; ++i)             // coalesced f32 reads
        tile[ty + i * 8][tx] = (f16)W[(size_t)(k0 + ty + i * 8) * DIMD + n0 + tx];
    __syncthreads();
#pragma unroll
    for (int i = 0; i < 4; ++i)             // coalesced f16 writes
        WT[(size_t)(n0 + ty + i * 8) * DIMD + k0 + tx] = tile[tx][ty + i * 8];
}

// ---------------------------------------------------------------------------
// GEMM tile core, software-pipelined (register double-buffered).
// One wave computes C[32 x 64] = A[32,K] * BT[64,K]^T
// A row-major f16 (lda = K), BT row-major f16 (rows = output cols, ldb = K)
// Accumulators: c[a] = cols a*16.., rows 0..15 ; c[4+a] = same cols, rows 16..31
// ---------------------------------------------------------------------------
static __device__ __forceinline__ void gemm_tile_32x64_pipe(
    const f16* __restrict__ A, const f16* __restrict__ BT,
    int K, int mrow, int n0, v8f c[8]) {
    const int lane = threadIdx.x & 31;
    const int half = lane >> 4;
    const int l16  = lane & 15;
    const f16* arow0 = A + (size_t)(mrow + l16) * K;
    const f16* arow1 = A + (size_t)(mrow + 16 + l16) * K;
    const f16* brow[4];
#pragma unroll
    for (int a = 0; a < 4; ++a)
        brow[a] = BT + (size_t)(n0 + a * 16 + l16) * K;

    auto ldA = [&](const f16* p, int kk) -> v16h {
        return cat16(*(const v8h*)(p + kk + half * 8),
                     *(const v8h*)(p + kk + 16 + half * 8));
    };
    auto ldB = [&](const f16* p, int kk) -> v16h {
        return cat16(*(const v8h*)(p + kk + half * 8), *(const v8h*)(p + kk + 16 + half * 8));
    };

    // prologue: load first k-block
    v16h a0 = ldA(arow0, 0), a1 = ldA(arow1, 0);
    v16h b0 = ldB(brow[0], 0), b1 = ldB(brow[1], 0);
    v16h b2 = ldB(brow[2], 0), b3 = ldB(brow[3], 0);

    for (int kk = 0; kk < K - 32; kk += 32) {
        const int kn = kk + 32;
        // speculative prefetch far ahead (harmless past end; no branch)
        __builtin_prefetch(arow0 + kk + 256, 0, 0);
        __builtin_prefetch(arow1 + kk + 256, 0, 0);
#pragma unroll
        for (int a = 0; a < 4; ++a)
            __builtin_prefetch(brow[a] + kk + 256, 0, 0);

        // issue next-iteration loads before current WMMAs (overlap)
        v16h a0n = ldA(arow0, kn), a1n = ldA(arow1, kn);
        v16h b0n = ldB(brow[0], kn), b1n = ldB(brow[1], kn);
        v16h b2n = ldB(brow[2], kn), b3n = ldB(brow[3], kn);

        c[0] = wmma16(a0, b0, c[0]);  c[4] = wmma16(a1, b0, c[4]);
        c[1] = wmma16(a0, b1, c[1]);  c[5] = wmma16(a1, b1, c[5]);
        c[2] = wmma16(a0, b2, c[2]);  c[6] = wmma16(a1, b2, c[6]);
        c[3] = wmma16(a0, b3, c[3]);  c[7] = wmma16(a1, b3, c[7]);

        a0 = a0n; a1 = a1n; b0 = b0n; b1 = b1n; b2 = b2n; b3 = b3n;
    }
    // epilogue k-block
    c[0] = wmma16(a0, b0, c[0]);  c[4] = wmma16(a1, b0, c[4]);
    c[1] = wmma16(a0, b1, c[1]);  c[5] = wmma16(a1, b1, c[5]);
    c[2] = wmma16(a0, b2, c[2]);  c[6] = wmma16(a1, b2, c[6]);
    c[3] = wmma16(a0, b3, c[3]);  c[7] = wmma16(a1, b3, c[7]);
}

// ---------------------------------------------------------------------------
// 3) QKV GEMM: xn[4096,2048] @ {wqT,wkT,wvT} -> q,k (f16, [ROWS,2048]),
//    v written directly transposed per head: vT [B,H,Dh,S] (f16)
//    4096 wave tiles (128 m-tiles of 32 x 32 n-tiles of 64) / 8 waves = 512 blk
// ---------------------------------------------------------------------------
__global__ __launch_bounds__(256) void gemm_qkv_kernel(
    const f16* __restrict__ xn,
    const f16* __restrict__ wqT, const f16* __restrict__ wkT,
    const f16* __restrict__ wvT,
    f16* __restrict__ q, f16* __restrict__ k, f16* __restrict__ vT) {
    const int wid = threadIdx.x >> 5;
    const int g   = blockIdx.x * 8 + wid;
    const int mrow = (g >> 5) << 5;             // 128 m-tiles of 32 rows
    const int n0   = (g & 31) << 6;             // 32 n-tiles of 64 cols
    const int sel  = blockIdx.y;

    const f16* BT = (sel == 0) ? wqT : (sel == 1) ? wkT : wvT;
    v8f c[8];
#pragma unroll
    for (int a = 0; a < 8; ++a)
#pragma unroll
        for (int i = 0; i < 8; ++i) c[a][i] = 0.f;

    gemm_tile_32x64_pipe(xn, BT, DIMD, mrow, n0, c);

    const int lane = threadIdx.x & 31;
    const int half = lane >> 4, l16 = lane & 15;
    if (sel < 2) {
        f16* dst = (sel == 0) ? q : k;
#pragma unroll
        for (int s = 0; s < 2; ++s)
#pragma unroll
            for (int a = 0; a < 4; ++a)
#pragma unroll
                for (int r = 0; r < 8; ++r) {
                    size_t row = mrow + s * 16 + r + 8 * half;
                    dst[row * DIMD + n0 + a * 16 + l16] = (f16)c[s * 4 + a][r];
                }
    } else {
#pragma unroll
        for (int s = 0; s < 2; ++s)
#pragma unroll
            for (int a = 0; a < 4; ++a)
#pragma unroll
                for (int r = 0; r < 8; ++r) {
                    int row = mrow + s * 16 + r + 8 * half;   // b*SEQ + t
                    int col = n0 + a * 16 + l16;              // h*64 + dh
                    int bb = row >> 11, ss = row & (SEQ - 1);
                    int hh = col >> 6,  dd = col & 63;
                    vT[(((size_t)bb * NHEADS + hh) * DHEAD + dd) * SEQ + ss] =
                        (f16)c[s * 4 + a][r];
                }
    }
}

// ---------------------------------------------------------------------------
// 4) Per-head LayerNorm of q and k (dim 64); fold 1/sqrt(64) into q.
//    One wave per head. grid = ROWS*NHEADS/8, block = 256 (8 waves).
// ---------------------------------------------------------------------------
__global__ __launch_bounds__(256) void qk_ln_kernel(f16* __restrict__ q,
                                                    f16* __restrict__ k,
                                                    const float* __restrict__ qw,
                                                    const float* __restrict__ kw) {
    const int wid  = threadIdx.x >> 5;
    const int lane = threadIdx.x & 31;
    const int bs   = blockIdx.x >> 2;           // row index in [0,ROWS)
    const int h    = (blockIdx.x & 3) * 8 + wid;
    const size_t base = (size_t)bs * DIMD + h * DHEAD;

#pragma unroll
    for (int which = 0; which < 2; ++which) {
        f16* buf = which ? k : q;
        const float* w = which ? kw : qw;
        const float scale = which ? 1.0f : 0.125f;   // 1/sqrt(DHEAD) folded into q
        float x0 = (float)buf[base + lane];
        float x1 = (float)buf[base + lane + 32];
        float s = x0 + x1, s2 = x0 * x0 + x1 * x1;
#pragma unroll
        for (int off = 16; off > 0; off >>= 1) {
            s  += __shfl_xor(s, off, 32);
            s2 += __shfl_xor(s2, off, 32);
        }
        const float mean = s * (1.0f / DHEAD);
        const float var  = s2 * (1.0f / DHEAD) - mean * mean;
        const float rstd = rsqrtf(var + 1e-5f);
        buf[base + lane]      = (f16)((x0 - mean) * rstd * w[lane] * scale);
        buf[base + lane + 32] = (f16)((x1 - mean) * rstd * w[lane + 32] * scale);
    }
}

// ---------------------------------------------------------------------------
// 5) Flash attention, fully pipelined. One wave = one (b,h,16-query tile).
//    Per 32-key block: V tiles + next-block K tiles are loaded FIRST, then
//    4 score WMMAs (operands from previous iteration's loads), softmax VALU
//    work hides the in-flight loads, P converts C->A layout via LDS
//    (ds_store_b16 -> s_wait_dscnt -> ds_load_b128), then 4 P*V WMMAs.
// ---------------------------------------------------------------------------
__global__ __launch_bounds__(256) void flash_kernel(const f16* __restrict__ q,
                                                    const f16* __restrict__ k,
                                                    const f16* __restrict__ vT,
                                                    f16* __restrict__ attn) {
    __shared__ f16 ldsP[8][16 * 32];
    const int wid  = threadIdx.x >> 5;
    const int lane = threadIdx.x & 31;
    const int half = lane >> 4, l16 = lane & 15;
    const int g  = blockIdx.x * 8 + wid;       // 8192 wave tiles
    const int bh = g >> 7;                     // 64 (b,h) pairs
    const int qt = g & 127;                    // 128 query tiles
    const int b = bh >> 5, h = bh & 31;
    const int q0 = qt << 4;
    f16* P = ldsP[wid];

    // Load Q A-tiles (16x64 split into two 16x32) once.
    const f16* qrow = q + (size_t)(b * SEQ + q0 + l16) * DIMD + h * DHEAD;
    v16h qa0 = cat16(*(const v8h*)(qrow + half * 8),
                     *(const v8h*)(qrow + 16 + half * 8));
    v16h qa1 = cat16(*(const v8h*)(qrow + 32 + half * 8),
                     *(const v8h*)(qrow + 48 + half * 8));

    v8f o[4];
    float m[8], l[8];
#pragma unroll
    for (int a = 0; a < 4; ++a)
#pragma unroll
        for (int i = 0; i < 8; ++i) o[a][i] = 0.f;
#pragma unroll
    for (int r = 0; r < 8; ++r) { m[r] = -1e30f; l[r] = 0.f; }

    const f16* kbase  = k + (size_t)b * SEQ * DIMD + h * DHEAD;
    const f16* vtbase = vT + ((size_t)b * NHEADS + h) * DHEAD * SEQ;

    auto loadK = [&](int kb_, v16h kt[4]) {
#pragma unroll
        for (int a = 0; a < 2; ++a) {
            const f16* krow = kbase + (size_t)(kb_ + a * 16 + l16) * DIMD;
            kt[a * 2 + 0] = cat16(*(const v8h*)(krow + half * 8),
                                  *(const v8h*)(krow + 16 + half * 8));
            kt[a * 2 + 1] = cat16(*(const v8h*)(krow + 32 + half * 8),
                                  *(const v8h*)(krow + 48 + half * 8));
        }
    };

    v16h kc[4], kn[4];
    loadK(0, kc);

    for (int kb = 0; kb < SEQ; kb += 32) {
        // ---- issue all loads for this iteration up front (overlap with math)
        v16h vv[4];
#pragma unroll
        for (int a = 0; a < 4; ++a) {
            const f16* vrow = vtbase + (size_t)(a * 16 + l16) * SEQ + kb + half * 8;
            vv[a] = cat16(*(const v8h*)vrow, *(const v8h*)(vrow + 16));
        }
        if (kb + 32 < SEQ) loadK(kb + 32, kn);

        // ---- scores: two 16x16 tiles over 32 keys (K-dim = 64 = 2 steps)
        v8f sc[2];
#pragma unroll
        for (int a = 0; a < 2; ++a) {
            v8f s;
#pragma unroll
            for (int i = 0; i < 8; ++i) s[i] = 0.f;
            s = wmma16(qa0, kc[a * 2 + 0], s);
            s = wmma16(qa1, kc[a * 2 + 1], s);
            sc[a] = s;
        }
        // ---- online softmax (rows live across 16-lane halves)
#pragma unroll
        for (int r = 0; r < 8; ++r) {
            float rowv = fmaxf(sc[0][r], sc[1][r]);
#pragma unroll
            for (int off = 8; off > 0; off >>= 1)
                rowv = fmaxf(rowv, __shfl_xor(rowv, off, 32));
            const float mnew  = fmaxf(m[r], rowv);
            const float alpha = __expf(m[r] - mnew);
            const float p0 = __expf(sc[0][r] - mnew);
            const float p1 = __expf(sc[1][r] - mnew);
            float rs = p0 + p1;
#pragma unroll
            for (int off = 8; off > 0; off >>= 1)
                rs += __shfl_xor(rs, off, 32);
            l[r] = l[r] * alpha + rs;
            m[r] = mnew;
            P[(r + 8 * half) * 32 + l16]      = (f16)p0;
            P[(r + 8 * half) * 32 + 16 + l16] = (f16)p1;
            o[0][r] *= alpha; o[1][r] *= alpha; o[2][r] *= alpha; o[3][r] *= alpha;
        }
        asm volatile("s_wait_dscnt 0" ::: "memory");
        // ---- reload P in A-layout (16x32 f16)
        v16h pv = cat16(*(const v8h*)&P[l16 * 32 + half * 8],
                        *(const v8h*)&P[l16 * 32 + 16 + half * 8]);
        // ---- P * V into 16x64 accumulator (V already in registers)
#pragma unroll
        for (int a = 0; a < 4; ++a)
            o[a] = wmma16(pv, vv[a], o[a]);
        if (kb + 32 < SEQ) {
#pragma unroll
            for (int i = 0; i < 4; ++i) kc[i] = kn[i];
        }
    }
    // ---- finalize: divide by l, store f16
#pragma unroll
    for (int r = 0; r < 8; ++r) {
        const float inv = 1.0f / l[r];
        const size_t row = (size_t)b * SEQ + q0 + r + 8 * half;
#pragma unroll
        for (int a = 0; a < 4; ++a)
            attn[row * DIMD + h * DHEAD + a * 16 + l16] = (f16)(o[a][r] * inv);
    }
}

// ---------------------------------------------------------------------------
// 6) Output GEMM: attn[4096,2048] f16 @ woT -> out fp32
// ---------------------------------------------------------------------------
__global__ __launch_bounds__(256) void gemm_out_kernel(const f16* __restrict__ A,
                                                       const f16* __restrict__ woT,
                                                       float* __restrict__ out) {
    const int wid = threadIdx.x >> 5;
    const int g   = blockIdx.x * 8 + wid;
    const int mrow = (g >> 5) << 5;
    const int n0   = (g & 31) << 6;
    v8f c[8];
#pragma unroll
    for (int a = 0; a < 8; ++a)
#pragma unroll
        for (int i = 0; i < 8; ++i) c[a][i] = 0.f;
    gemm_tile_32x64_pipe(A, woT, DIMD, mrow, n0, c);
    const int lane = threadIdx.x & 31;
    const int half = lane >> 4, l16 = lane & 15;
#pragma unroll
    for (int s = 0; s < 2; ++s)
#pragma unroll
        for (int a = 0; a < 4; ++a)
#pragma unroll
            for (int r = 0; r < 8; ++r) {
                size_t row = mrow + s * 16 + r + 8 * half;
                out[row * DIMD + n0 + a * 16 + l16] = c[s * 4 + a][r];
            }
}

// ---------------------------------------------------------------------------
extern "C" void kernel_launch(void* const* d_in, const int* in_sizes, int n_in,
                              void* d_out, int out_size, void* d_ws, size_t ws_size,
                              hipStream_t stream) {
    const float* x      = (const float*)d_in[0];
    const float* norm_w = (const float*)d_in[1];
    const float* wq     = (const float*)d_in[2];
    const float* wk     = (const float*)d_in[3];
    const float* wv     = (const float*)d_in[4];
    const float* qn_w   = (const float*)d_in[5];
    const float* kn_w   = (const float*)d_in[6];
    const float* wo     = (const float*)d_in[7];
    float* out = (float*)d_out;

    char* ws = (char*)d_ws;
    size_t off = 0;
    f16* xn   = (f16*)(ws + off); off += (size_t)ROWS * DIMD * 2;   // 16 MiB
    f16* wqT  = (f16*)(ws + off); off += (size_t)DIMD * DIMD * 2;   //  8 MiB
    f16* wkT  = (f16*)(ws + off); off += (size_t)DIMD * DIMD * 2;
    f16* wvT  = (f16*)(ws + off); off += (size_t)DIMD * DIMD * 2;
    f16* woT  = (f16*)(ws + off); off += (size_t)DIMD * DIMD * 2;
    f16* qb   = (f16*)(ws + off); off += (size_t)ROWS * DIMD * 2;   // 16 MiB
    f16* kb   = (f16*)(ws + off); off += (size_t)ROWS * DIMD * 2;   // 16 MiB
    f16* vT   = (f16*)(ws + off); off += (size_t)ROWS * DIMD * 2;   // 16 MiB
    f16* attn = (f16*)(ws + off); off += (size_t)ROWS * DIMD * 2;   // 16 MiB

    ln_x_kernel<<<ROWS, 256, 0, stream>>>(x, norm_w, xn);

    dim3 tgrid(DIMD / 32, DIMD / 32);
    wconv_kernel<<<tgrid, 256, 0, stream>>>(wq, wqT);
    wconv_kernel<<<tgrid, 256, 0, stream>>>(wk, wkT);
    wconv_kernel<<<tgrid, 256, 0, stream>>>(wv, wvT);
    wconv_kernel<<<tgrid, 256, 0, stream>>>(wo, woT);

    // (ROWS/32) * (DIMD/64) = 4096 wave tiles / 8 waves per block = 512 blocks
    gemm_qkv_kernel<<<dim3(512, 3), 256, 0, stream>>>(xn, wqT, wkT, wvT, qb, kb, vT);

    qk_ln_kernel<<<(ROWS * NHEADS) / 8, 256, 0, stream>>>(qb, kb, qn_w, kn_w);

    // 64 (b,h) * 128 qtiles = 8192 wave tiles / 8 per block
    flash_kernel<<<1024, 256, 0, stream>>>(qb, kb, vT, attn);

    gemm_out_kernel<<<512, 256, 0, stream>>>(attn, woT, out);
}
